// RSCNN_SSN_54606214201520
// MI455X (gfx1250) — compile-verified, matrix operand order checked by reading, not run on hardware
//
#include <hip/hip_runtime.h>
#include <hip/hip_bf16.h>

typedef __attribute__((ext_vector_type(16))) _Float16 v16h;
typedef __attribute__((ext_vector_type(8)))  _Float16 v8h;
typedef __attribute__((ext_vector_type(8)))  float    v8f;

#define BN_EPS 1e-5f
#define BATCH 16

// ---------------------------------------------------------------------------
// Farthest point sampling: one block per batch. Point cloud cached in LDS
// (SoA), distances in registers, wave32 shuffle argmax + cross-wave combine
// done redundantly by all threads => 2 barriers per sequential iteration.
// ---------------------------------------------------------------------------
__global__ __launch_bounds__(256)
void fps_kernel(const float* __restrict__ xyz, int n, int npoint,
                int* __restrict__ outIdx) {
    extern __shared__ float sxyz[];           // 3*n floats: sx | sy | sz
    float* sx = sxyz;
    float* sy = sxyz + n;
    float* sz = sxyz + 2 * n;
    __shared__ float wval[8];
    __shared__ int   widx[8];

    const int T = 256;
    int b = blockIdx.x;
    int tid = threadIdx.x;
    const float* xb = xyz + (size_t)b * n * 3;
    for (int p = tid; p < n; p += T) {
        sx[p] = xb[p * 3 + 0];
        sy[p] = xb[p * 3 + 1];
        sz[p] = xb[p * 3 + 2];
    }
    __syncthreads();

    int cnt = (n + T - 1) / T;
    float d[16];
    for (int i = 0; i < 16; ++i) d[i] = 1e10f;

    int last = 0;
    for (int it = 0; it < npoint; ++it) {
        if (tid == 0) outIdx[(size_t)b * npoint + it] = last;
        float lx = sx[last], ly = sy[last], lz = sz[last];
        float best = -1.0f; int bi = 0x7fffffff;
        for (int i = 0; i < cnt; ++i) {
            int p = tid + i * T;
            if (p < n) {
                float dx = sx[p] - lx;
                float dy = sy[p] - ly;
                float dz = sz[p] - lz;
                float dd = dx * dx + dy * dy + dz * dz;
                if (dd < d[i]) d[i] = dd;
                if (d[i] > best) { best = d[i]; bi = p; }
            }
        }
        // wave32 shuffle argmax (first-index tie-break)
#pragma unroll
        for (int off = 16; off > 0; off >>= 1) {
            float ov = __shfl_xor(best, off, 32);
            int   oi = __shfl_xor(bi, off, 32);
            if (ov > best || (ov == best && oi < bi)) { best = ov; bi = oi; }
        }
        if ((tid & 31) == 0) { wval[tid >> 5] = best; widx[tid >> 5] = bi; }
        __syncthreads();
        float fb = wval[0]; int fi = widx[0];
#pragma unroll
        for (int w = 1; w < 8; ++w) {
            float ov = wval[w]; int oi = widx[w];
            if (ov > fb || (ov == fb && oi < fi)) { fb = ov; fi = oi; }
        }
        last = fi;           // uniform across the block
        __syncthreads();     // protect wval/widx before next overwrite
    }
}

// ---------------------------------------------------------------------------
__global__ void gather_xyz_kernel(const float* __restrict__ xyz,
                                  const int* __restrict__ idx,
                                  float* __restrict__ out, int n, int S) {
    int t = blockIdx.x * blockDim.x + threadIdx.x;
    int b = blockIdx.y;
    if (t >= S) return;
    int id = idx[(size_t)b * S + t];
    const float* src = xyz + ((size_t)b * n + id) * 3;
    float* dst = out + ((size_t)b * S + t) * 3;
    dst[0] = src[0]; dst[1] = src[1]; dst[2] = src[2];
}

// ---------------------------------------------------------------------------
// Ball query with LDS-cached point cloud: ordered scan collects first ns
// indices with d2 < r2, pads with first found (matches reference semantics).
// ---------------------------------------------------------------------------
__global__ __launch_bounds__(256)
void ball_query_kernel(const float* __restrict__ xyz,
                       const float* __restrict__ new_xyz,
                       int* __restrict__ idx,
                       int n, int S, int ns, float r2) {
    extern __shared__ float sbq[];            // 3*n floats
    float* sx = sbq;
    float* sy = sbq + n;
    float* sz = sbq + 2 * n;
    int b = blockIdx.y;
    const float* xb = xyz + (size_t)b * n * 3;
    for (int p = threadIdx.x; p < n; p += blockDim.x) {
        sx[p] = xb[p * 3 + 0];
        sy[p] = xb[p * 3 + 1];
        sz[p] = xb[p * 3 + 2];
    }
    __syncthreads();

    int t = blockIdx.x * blockDim.x + threadIdx.x;
    if (t >= S) return;
    const float* c = new_xyz + ((size_t)b * S + t) * 3;
    float cx = c[0], cy = c[1], cz = c[2];
    int* o = idx + ((size_t)b * S + t) * ns;
    int cnt = 0;
    for (int p = 0; p < n && cnt < ns; ++p) {
        float dx = sx[p] - cx;
        float dy = sy[p] - cy;
        float dz = sz[p] - cz;
        if (dx * dx + dy * dy + dz * dz < r2) o[cnt++] = p;
    }
    int pad = (cnt > 0) ? o[0] : (n - 1);
    for (int j = cnt; j < ns; ++j) o[j] = pad;
}

// ---------------------------------------------------------------------------
// Build relation features H (10ch in 32-row padded buffer) and x-features
// X (3+F ch in XCpad-row padded buffer), f16, layout (B, Cpad, P), pad rows 0.
// H = [ |rel| , raw(first-neighbor) , raw , rel ]
// ---------------------------------------------------------------------------
__global__ void group_kernel(const float* __restrict__ xyz,
                             const float* __restrict__ new_xyz,
                             const int* __restrict__ idx,
                             const _Float16* __restrict__ feats,  // (B,F,n) or null
                             _Float16* __restrict__ H,            // (B,32,P)
                             _Float16* __restrict__ X,            // (B,XCpad,P)
                             int n, int S, int ns, int F, int XCpad) {
    int t = blockIdx.x * blockDim.x + threadIdx.x;
    int b = blockIdx.y;
    int P = S * ns;
    if (t >= P) return;
    int s = t / ns, j = t % ns;
    const int* idr = idx + ((size_t)b * S + s) * ns;
    int id = idr[j];
    int id0 = idr[0];
    const float* pp = xyz + ((size_t)b * n + id) * 3;
    const float* q0 = xyz + ((size_t)b * n + id0) * 3;
    const float* cc = new_xyz + ((size_t)b * S + s) * 3;
    float px = pp[0], py = pp[1], pz = pp[2];
    float rx = px - cc[0], ry = py - cc[1], rz = pz - cc[2];
    float s2 = rx * rx + ry * ry + rz * rz;
    float nrm = (s2 == 0.0f) ? 0.0f : sqrtf(s2);
    _Float16* Hb = H + (size_t)b * 32 * P + t;
    Hb[0 * (size_t)P] = (_Float16)nrm;
    Hb[1 * (size_t)P] = (_Float16)q0[0];
    Hb[2 * (size_t)P] = (_Float16)q0[1];
    Hb[3 * (size_t)P] = (_Float16)q0[2];
    Hb[4 * (size_t)P] = (_Float16)px;
    Hb[5 * (size_t)P] = (_Float16)py;
    Hb[6 * (size_t)P] = (_Float16)pz;
    Hb[7 * (size_t)P] = (_Float16)rx;
    Hb[8 * (size_t)P] = (_Float16)ry;
    Hb[9 * (size_t)P] = (_Float16)rz;
    for (int c = 10; c < 32; ++c) Hb[(size_t)c * P] = (_Float16)0.0f;

    _Float16* Xb = X + (size_t)b * XCpad * P + t;
    Xb[0]             = (_Float16)rx;
    Xb[(size_t)P]     = (_Float16)ry;
    Xb[2 * (size_t)P] = (_Float16)rz;
    for (int f = 0; f < F; ++f)
        Xb[(size_t)(3 + f) * P] = feats[((size_t)b * F + f) * n + id];
    for (int c = 3 + F; c < XCpad; ++c) Xb[(size_t)c * P] = (_Float16)0.0f;
}

// ---------------------------------------------------------------------------
// Weight prep: f32 (O,C) -> zero-padded f16 (Opad,Cpad) + zero-padded bias.
// ---------------------------------------------------------------------------
__global__ void prep_weights_kernel(const float* __restrict__ W,
                                    const float* __restrict__ bias,
                                    _Float16* __restrict__ Wh,
                                    float* __restrict__ biasPad,
                                    int O, int C, int Opad, int Cpad) {
    int t = blockIdx.x * blockDim.x + threadIdx.x;
    int total = Opad * Cpad;
    if (t < total) {
        int o = t / Cpad, c = t % Cpad;
        Wh[t] = (o < O && c < C) ? (_Float16)W[(size_t)o * C + c] : (_Float16)0.0f;
    }
    if (t < Opad) biasPad[t] = (t < O) ? bias[t] : 0.0f;
}

// ---------------------------------------------------------------------------
// conv1x1 as WMMA GEMM on padded operands.
// Block = 2 waves sharing one 32-column N panel. The 32(K)x32(N) f16 B tile
// is staged in LDS with GLOBAL_LOAD_ASYNC_TO_LDS_B128 (ASYNCcnt), double-
// buffered so the next tile streams in while this tile feeds the WMMAs.
// A fragments = two b128 loads of consecutive K-halves per ISA 7.12.2.
// All control flow is scalar-uniform; EXEC all-1s at the WMMAs.
// ---------------------------------------------------------------------------
__global__ __launch_bounds__(64)
void gemm_wmma_kernel(const _Float16* __restrict__ Wh,
                      const float* __restrict__ biasPad,
                      const _Float16* __restrict__ X, _Float16* __restrict__ Y,
                      int Opad, int Cpad, int P) {
    __shared__ _Float16 Btile[2][32 * 32];    // [k][n] row-major, 2KB each
    int tid  = threadIdx.x;                   // 0..63
    int wave = tid >> 5;
    int lane = tid & 31;
    int mn = lane & 15;
    int hi8 = (lane >> 4) << 3;               // 0 / 8
    int n0 = blockIdx.x * 32;
    int mBase = (blockIdx.y * 2 + wave) * 16;
    const _Float16* Xb = X + (size_t)blockIdx.z * Cpad * P;
    const _Float16* Arow = Wh + (size_t)(mBase + mn) * Cpad + hi8;

    // async-copy mapping: thread -> (row r0 = tid/4, 16-byte chunk ch = tid%4)
    int r0 = tid >> 2, ch = tid & 3;
    unsigned ldsBase[2];
    ldsBase[0] = (unsigned)(size_t)(&Btile[0][0]);
    ldsBase[1] = (unsigned)(size_t)(&Btile[1][0]);
    unsigned ldsOff = (unsigned)((r0 * 32 + ch * 8) * 2);

    auto issueTile = [&](int buf, int k0) {
        // rows r0 and r0+16 of the 32-row tile, 16 bytes per lane each
        const _Float16* g0 = Xb + (size_t)(k0 + r0) * P + n0 + ch * 8;
        const _Float16* g1 = g0 + (size_t)16 * P;
        unsigned l0 = ldsBase[buf] + ldsOff;
        unsigned l1 = l0 + 16 * 32 * 2;
        asm volatile("global_load_async_to_lds_b128 %0, %1, off"
                     :: "v"(l0), "v"(g0) : "memory");
        asm volatile("global_load_async_to_lds_b128 %0, %1, off"
                     :: "v"(l1), "v"(g1) : "memory");
    };

    v8f acc0 = {}, acc1 = {};
    union V16H { v16h v; unsigned u[8]; };

    issueTile(0, 0);
    for (int k0 = 0; k0 < Cpad; k0 += 32) {
        int cur = (k0 >> 5) & 1;
        asm volatile("s_wait_asynccnt 0x0" ::: "memory");
        __syncthreads();                       // tile[cur] visible to all waves
        if (k0 + 32 < Cpad) issueTile(cur ^ 1, k0 + 32);

        // A: lane fragment = halves [k0+hi8 .. +7] and [k0+16+hi8 .. +7]
        v8h alo = *(const v8h*)(Arow + k0);
        v8h ahi = *(const v8h*)(Arow + k0 + 16);
        v16h a;
#pragma unroll
        for (int i = 0; i < 8; ++i) { a[i] = alo[i]; a[i + 8] = ahi[i]; }

        // B fragments from LDS [k][n] tile, packed into full 32-bit regs
        const _Float16* Bl = &Btile[cur][0];
        V16H B0, B1;
#pragma unroll
        for (int w = 0; w < 8; ++w) {
            int i = 2 * w;
            int kl = ((i < 8) ? i : i + 8) + hi8;   // element i ; i+1 is kl+1
            unsigned lo0 = *(const unsigned short*)(Bl + kl * 32 + mn);
            unsigned hi0 = *(const unsigned short*)(Bl + (kl + 1) * 32 + mn);
            unsigned lo1 = *(const unsigned short*)(Bl + kl * 32 + 16 + mn);
            unsigned hi1 = *(const unsigned short*)(Bl + (kl + 1) * 32 + 16 + mn);
            B0.u[w] = lo0 | (hi0 << 16);
            B1.u[w] = lo1 | (hi1 << 16);
        }
        acc0 = __builtin_amdgcn_wmma_f32_16x16x32_f16(
            false, a, false, B0.v, (short)0, acc0, false, false);
        acc1 = __builtin_amdgcn_wmma_f32_16x16x32_f16(
            false, a, false, B1.v, (short)0, acc1, false, false);
        __syncthreads();                       // reads done before buf reuse
    }
    _Float16* Yb = Y + (size_t)blockIdx.z * Opad * P;
#pragma unroll
    for (int v = 0; v < 8; ++v) {
        int m = mBase + v + hi8;
        float bv = biasPad[m];
        Yb[(size_t)m * P + n0 + mn]      = (_Float16)(acc0[v] + bv);
        Yb[(size_t)m * P + n0 + 16 + mn] = (_Float16)(acc1[v] + bv);
    }
}

// ---------------------------------------------------------------------------
// BN stats: per-channel sum / sumsq over (B, P); shuffle-reduce + one barrier.
// ---------------------------------------------------------------------------
__global__ __launch_bounds__(256)
void stats_kernel(const _Float16* __restrict__ X, float* __restrict__ stat,
                  int Cpad, int P) {
    int c = blockIdx.x, b = blockIdx.y;
    const _Float16* p = X + ((size_t)b * Cpad + c) * P;
    float s = 0.0f, s2 = 0.0f;
    for (int i = threadIdx.x; i < P; i += 256) {
        float v = (float)p[i];
        s += v; s2 += v * v;
    }
#pragma unroll
    for (int off = 16; off > 0; off >>= 1) {
        s  += __shfl_xor(s, off, 32);
        s2 += __shfl_xor(s2, off, 32);
    }
    __shared__ float wa[8], wb[8];
    if ((threadIdx.x & 31) == 0) { wa[threadIdx.x >> 5] = s; wb[threadIdx.x >> 5] = s2; }
    __syncthreads();
    if (threadIdx.x == 0) {
        float t = 0.0f, t2 = 0.0f;
#pragma unroll
        for (int w = 0; w < 8; ++w) { t += wa[w]; t2 += wb[w]; }
        atomicAdd(&stat[2 * c + 0], t);
        atomicAdd(&stat[2 * c + 1], t2);
    }
}

// ---------------------------------------------------------------------------
// In-place BN + ReLU on the C real channels (pad rows stay zero).
// ---------------------------------------------------------------------------
__global__ __launch_bounds__(256)
void bn_relu_kernel(_Float16* __restrict__ X, const float* __restrict__ stat,
                    const float* __restrict__ g, const float* __restrict__ bt,
                    int Cpad, int P, float invCount) {
    int c = blockIdx.x, b = blockIdx.y;
    float m = stat[2 * c] * invCount;
    float var = stat[2 * c + 1] * invCount - m * m;
    float sc = g[c] * rsqrtf(var + BN_EPS);
    float bb = bt[c] - m * sc;
    _Float16* p = X + ((size_t)b * Cpad + c) * P;
    for (int i = threadIdx.x; i < P; i += 256) {
        float v = (float)p[i] * sc + bb;
        p[i] = (_Float16)fmaxf(v, 0.0f);
    }
}

// ---------------------------------------------------------------------------
__global__ void mul_kernel(_Float16* __restrict__ A, const _Float16* __restrict__ B,
                           size_t total) {
    size_t i = (size_t)blockIdx.x * blockDim.x + threadIdx.x;
    if (i < total) A[i] = (_Float16)((float)A[i] * (float)B[i]);
}

// ---------------------------------------------------------------------------
// BN + ReLU then max over neighborhood (ns fastest). Writes Cpad rows; rows
// c >= C get explicit zeros so the pooled buffer is valid padded GEMM input.
// ---------------------------------------------------------------------------
template <typename OutT>
__global__ void bn_relu_max_kernel(const _Float16* __restrict__ X,
                                   const float* __restrict__ stat,
                                   const float* __restrict__ g,
                                   const float* __restrict__ bt,
                                   OutT* __restrict__ out,
                                   int C, int Cpad, int S, int ns,
                                   float invCount) {
    int t = blockIdx.x * blockDim.x + threadIdx.x;
    int b = blockIdx.y;
    if (t >= Cpad * S) return;
    int c = t / S, s = t % S;
    if (c >= C) {
        out[((size_t)b * Cpad + c) * S + s] = (OutT)0.0f;
        return;
    }
    float m = stat[2 * c] * invCount;
    float var = stat[2 * c + 1] * invCount - m * m;
    float sc = g[c] * rsqrtf(var + BN_EPS);
    float bb = bt[c] - m * sc;
    const _Float16* p = X + (((size_t)b * Cpad + c) * S + s) * ns;
    float best = 0.0f;  // relu outputs are >= 0
    for (int j = 0; j < ns; ++j) {
        float v = fmaxf((float)p[j] * sc + bb, 0.0f);
        if (v > best) best = v;
    }
    out[((size_t)b * Cpad + c) * S + s] = (OutT)best;
}

// ---------------------------------------------------------------------------
// Module-3 concat into 544-row padded buffer (rows >= 515 zeroed).
// ---------------------------------------------------------------------------
__global__ void concat3_kernel(const float* __restrict__ xyz2,    // (B,128,3)
                               const _Float16* __restrict__ f2,   // (B,512,128)
                               _Float16* __restrict__ G) {        // (B,544,128)
    int t = blockIdx.x * blockDim.x + threadIdx.x;
    int b = blockIdx.y;
    const int P = 128, CHP = 544, F = 512;
    if (t >= CHP * P) return;
    int c = t / P, p = t % P;
    float v = 0.0f;
    if (c < 3)        v = xyz2[((size_t)b * P + p) * 3 + c];
    else if (c < 515) v = (float)f2[((size_t)b * F + (c - 3)) * P + p];
    G[(size_t)b * CHP * P + t] = (_Float16)v;
}

// ---------------------------------------------------------------------------
// Host driver
// ---------------------------------------------------------------------------
enum {
    IN_PC = 0,
    P_M1_W_MAP1, P_M1_B_MAP1, P_M1_BN_MAP_G, P_M1_BN_MAP_B,
    P_M1_W_MAP2, P_M1_B_MAP2,
    P_M1_W_XYZ, P_M1_B_XYZ, P_M1_BN_XYZ_G, P_M1_BN_XYZ_B,
    P_M1_BN_RS_G, P_M1_BN_RS_B,
    P_M1_W_CR, P_M1_B_CR, P_M1_BN_CR_G, P_M1_BN_CR_B,
    P_M2_W_MAP1, P_M2_B_MAP1, P_M2_BN_MAP_G, P_M2_BN_MAP_B,
    P_M2_W_MAP2, P_M2_B_MAP2, P_M2_BN_RS_G, P_M2_BN_RS_B,
    P_M2_W_CR, P_M2_B_CR, P_M2_BN_CR_G, P_M2_BN_CR_B,
    P_M3_W, P_M3_B, P_M3_BN_G, P_M3_BN_B
};

extern "C" void kernel_launch(void* const* d_in, const int* in_sizes, int n_in,
                              void* d_out, int out_size, void* d_ws, size_t ws_size,
                              hipStream_t stream) {
    (void)in_sizes; (void)n_in; (void)out_size; (void)ws_size;
    const int B = BATCH, N = 4096;
    const float* pc = (const float*)d_in[IN_PC];
    auto F32 = [&](int i) { return (const float*)d_in[i]; };

    // --- bump allocator over workspace ---
    char* ws = (char*)d_ws;
    size_t off = 0;
    auto alloc = [&](size_t bytes) -> void* {
        void* p = ws + off;
        off += (bytes + 255) & ~(size_t)255;
        return p;
    };

    float* stat      = (float*)alloc(2 * 1024 * sizeof(float));
    int*   fpsidx1   = (int*)alloc((size_t)B * 512 * sizeof(int));
    int*   fpsidx2   = (int*)alloc((size_t)B * 128 * sizeof(int));
    float* new_xyz1  = (float*)alloc((size_t)B * 512 * 3 * sizeof(float));
    float* new_xyz2  = (float*)alloc((size_t)B * 128 * 3 * sizeof(float));
    int*   idx1      = (int*)alloc((size_t)B * 512 * 48 * sizeof(int));
    int*   idx2      = (int*)alloc((size_t)B * 128 * 64 * sizeof(int));
    const int P1 = 512 * 48;      // module-1 positions per batch
    const int P2 = 128 * 64;      // module-2 positions per batch
    // Channel-padded f16 activation buffers (pad rows kept zero by writers)
    _Float16* H1   = (_Float16*)alloc((size_t)B * 32   * P1 * 2);
    _Float16* X1   = (_Float16*)alloc((size_t)B * 32   * P1 * 2);
    _Float16* X16  = (_Float16*)alloc((size_t)B * 32   * P1 * 2);
    _Float16* T64  = (_Float16*)alloc((size_t)B * 64   * P1 * 2);
    _Float16* H2b  = (_Float16*)alloc((size_t)B * 32   * P1 * 2);
    _Float16* pool1= (_Float16*)alloc((size_t)B * 32   * 512 * 2);
    _Float16* F1   = (_Float16*)alloc((size_t)B * 128  * 512 * 2);
    _Float16* H2m  = (_Float16*)alloc((size_t)B * 32   * P2 * 2);
    _Float16* X2   = (_Float16*)alloc((size_t)B * 160  * P2 * 2);
    _Float16* T128 = (_Float16*)alloc((size_t)B * 128  * P2 * 2);
    _Float16* HH   = (_Float16*)alloc((size_t)B * 160  * P2 * 2);
    _Float16* pool2= (_Float16*)alloc((size_t)B * 160  * 128 * 2);
    _Float16* F2   = (_Float16*)alloc((size_t)B * 512  * 128 * 2);
    _Float16* G3   = (_Float16*)alloc((size_t)B * 544  * 128 * 2);
    _Float16* Y3   = (_Float16*)alloc((size_t)B * 1024 * 128 * 2);

    // Padded f16 weights + padded bias
    struct WSlot { _Float16* w; float* b; };
    auto wslot = [&](int Opad, int Cpad) -> WSlot {
        WSlot s;
        s.w = (_Float16*)alloc((size_t)Opad * Cpad * 2);
        s.b = (float*)alloc((size_t)Opad * sizeof(float));
        return s;
    };
    WSlot w_xyz  = wslot(32, 32);      // 16x3
    WSlot w_m1a  = wslot(64, 32);      // 64x10
    WSlot w_m1b  = wslot(32, 64);      // 16x64
    WSlot w_cr1  = wslot(128, 32);     // 128x16
    WSlot w_m2a  = wslot(128, 32);     // 128x10
    WSlot w_m2b  = wslot(160, 128);    // 131x128
    WSlot w_cr2  = wslot(512, 160);    // 512x131
    WSlot w_m3   = wslot(1024, 544);   // 1024x515

    auto prep = [&](int wi, int bi, WSlot s, int O, int C, int Opad, int Cpad) {
        int total = Opad * Cpad;
        prep_weights_kernel<<<(total + 255) / 256, 256, 0, stream>>>(
            F32(wi), F32(bi), s.w, s.b, O, C, Opad, Cpad);
    };
    prep(P_M1_W_XYZ,  P_M1_B_XYZ,  w_xyz, 16, 3,    32, 32);
    prep(P_M1_W_MAP1, P_M1_B_MAP1, w_m1a, 64, 10,   64, 32);
    prep(P_M1_W_MAP2, P_M1_B_MAP2, w_m1b, 16, 64,   32, 64);
    prep(P_M1_W_CR,   P_M1_B_CR,   w_cr1, 128, 16,  128, 32);
    prep(P_M2_W_MAP1, P_M2_B_MAP1, w_m2a, 128, 10,  128, 32);
    prep(P_M2_W_MAP2, P_M2_B_MAP2, w_m2b, 131, 128, 160, 128);
    prep(P_M2_W_CR,   P_M2_B_CR,   w_cr2, 512, 131, 512, 160);
    prep(P_M3_W,      P_M3_B,      w_m3,  1024, 515, 1024, 544);

    auto gemm = [&](WSlot s, const _Float16* X, _Float16* Y,
                    int Opad, int Cpad, int P) {
        dim3 grid(P / 32, Opad / 32, B);
        gemm_wmma_kernel<<<grid, 64, 0, stream>>>(s.w, s.b, X, Y, Opad, Cpad, P);
    };
    auto bn = [&](_Float16* X, const float* g, const float* bt,
                  int C, int Cpad, int P) {
        hipMemsetAsync(stat, 0, 2 * (size_t)C * sizeof(float), stream);
        stats_kernel<<<dim3(C, B), 256, 0, stream>>>(X, stat, Cpad, P);
        bn_relu_kernel<<<dim3(C, B), 256, 0, stream>>>(
            X, stat, g, bt, Cpad, P, 1.0f / ((float)B * P));
    };

    // ============================ Module 1 =================================
    fps_kernel<<<B, 256, 3 * N * sizeof(float), stream>>>(pc, N, 512, fpsidx1);
    gather_xyz_kernel<<<dim3(2, B), 256, 0, stream>>>(pc, fpsidx1, new_xyz1, N, 512);
    ball_query_kernel<<<dim3(2, B), 256, 3 * N * sizeof(float), stream>>>(
        pc, new_xyz1, idx1, N, 512, 48, 0.23f * 0.23f);
    group_kernel<<<dim3((P1 + 255) / 256, B), 256, 0, stream>>>(
        pc, new_xyz1, idx1, nullptr, H1, X1, N, 512, 48, 0, 32);

    gemm(w_xyz, X1, X16, 32, 32, P1);
    bn(X16, F32(P_M1_BN_XYZ_G), F32(P_M1_BN_XYZ_B), 16, 32, P1);

    gemm(w_m1a, H1, T64, 64, 32, P1);
    bn(T64, F32(P_M1_BN_MAP_G), F32(P_M1_BN_MAP_B), 64, 64, P1);
    gemm(w_m1b, T64, H2b, 32, 64, P1);

    {
        size_t tot = (size_t)B * 32 * P1;
        mul_kernel<<<(unsigned)((tot + 255) / 256), 256, 0, stream>>>(H2b, X16, tot);
    }
    hipMemsetAsync(stat, 0, 2 * 16 * sizeof(float), stream);
    stats_kernel<<<dim3(16, B), 256, 0, stream>>>(H2b, stat, 32, P1);
    bn_relu_max_kernel<_Float16><<<dim3((32 * 512 + 255) / 256, B), 256, 0, stream>>>(
        H2b, stat, F32(P_M1_BN_RS_G), F32(P_M1_BN_RS_B), pool1,
        16, 32, 512, 48, 1.0f / ((float)B * P1));

    gemm(w_cr1, pool1, F1, 128, 32, 512);
    bn(F1, F32(P_M1_BN_CR_G), F32(P_M1_BN_CR_B), 128, 128, 512);

    // ============================ Module 2 =================================
    fps_kernel<<<B, 256, 3 * 512 * sizeof(float), stream>>>(new_xyz1, 512, 128, fpsidx2);
    gather_xyz_kernel<<<dim3(1, B), 256, 0, stream>>>(new_xyz1, fpsidx2, new_xyz2, 512, 128);
    ball_query_kernel<<<dim3(1, B), 256, 3 * 512 * sizeof(float), stream>>>(
        new_xyz1, new_xyz2, idx2, 512, 128, 64, 0.32f * 0.32f);
    group_kernel<<<dim3((P2 + 255) / 256, B), 256, 0, stream>>>(
        new_xyz1, new_xyz2, idx2, F1, H2m, X2, 512, 128, 64, 128, 160);

    gemm(w_m2a, H2m, T128, 128, 32, P2);
    bn(T128, F32(P_M2_BN_MAP_G), F32(P_M2_BN_MAP_B), 128, 128, P2);
    gemm(w_m2b, T128, HH, 160, 128, P2);

    {
        size_t tot = (size_t)B * 160 * P2;
        mul_kernel<<<(unsigned)((tot + 255) / 256), 256, 0, stream>>>(HH, X2, tot);
    }
    hipMemsetAsync(stat, 0, 2 * 131 * sizeof(float), stream);
    stats_kernel<<<dim3(131, B), 256, 0, stream>>>(HH, stat, 160, P2);
    bn_relu_max_kernel<_Float16><<<dim3((160 * 128 + 255) / 256, B), 256, 0, stream>>>(
        HH, stat, F32(P_M2_BN_RS_G), F32(P_M2_BN_RS_B), pool2,
        131, 160, 128, 64, 1.0f / ((float)B * P2));

    gemm(w_cr2, pool2, F2, 512, 160, 128);
    bn(F2, F32(P_M2_BN_CR_G), F32(P_M2_BN_CR_B), 512, 512, 128);

    // ============================ Module 3 =================================
    concat3_kernel<<<dim3((544 * 128 + 255) / 256, B), 256, 0, stream>>>(
        new_xyz2, F2, G3);
    gemm(w_m3, G3, Y3, 1024, 544, 128);
    hipMemsetAsync(stat, 0, 2 * 1024 * sizeof(float), stream);
    stats_kernel<<<dim3(1024, B), 256, 0, stream>>>(Y3, stat, 1024, 128);
    bn_relu_max_kernel<float><<<dim3((1024 + 255) / 256, B), 256, 0, stream>>>(
        Y3, stat, F32(P_M3_BN_G), F32(P_M3_BN_B), (float*)d_out,
        1024, 1024, 1, 128, 1.0f / ((float)B * 128));
}
